// Generator_19911468384444
// MI455X (gfx1250) — matrix operand order repaired and checked by reading, not compiled, not run on hardware
//
#include <hip/hip_runtime.h>
#include <hip/hip_bf16.h>
#include <math.h>

// ---------------- model dims ----------------
#define VDIM 390
#define EDIM 256
#define HDIM 512
#define BDIM 64
#define TDIM 512
#define START_TOK 388
#define G3H 1536           // 3*H
#define VPAD 416           // V padded to multiple of 32 for WMMA K loop

typedef __bf16 bf16_t;
typedef __attribute__((ext_vector_type(16))) __bf16 v16bf;
typedef __attribute__((ext_vector_type(8)))  float  v8f;

// ---------------- WMMA helpers (CDNA5 bf16 16x16x32, fp32 acc) ----------------
__device__ __forceinline__ v8f wmma_bf16(v16bf a, v16bf b, v8f c) {
  return __builtin_amdgcn_wmma_f32_16x16x32_bf16(
      /*neg_a=*/false, a, /*neg_b=*/false, b,
      /*c_mod=*/(short)0, c, /*reuse_a=*/false, /*reuse_b=*/false);
}

// A-matrix fragment 16x32 bf16, act row-major [M x K], lda = K.
// Lane layout (ISA 7.12.2): lanes 0-15 hold K = k0+{0..7, 16..23},
// lanes 16-31 hold K = k0+{8..15, 24..31}; row M = lane&15.
__device__ __forceinline__ v16bf load_afrag(const bf16_t* __restrict__ A,
                                            int lda, int m0, int k0, int lane) {
  int half = (lane >> 4) & 1;
  const bf16_t* p = A + (size_t)(m0 + (lane & 15)) * lda + k0 + half * 8;
  v16bf a;
#pragma unroll
  for (int i = 0; i < 8; ++i) { a[i] = p[i]; a[i + 8] = p[16 + i]; }
  return a;
}

// B-matrix fragment 32x16 bf16 where B(k,n) = W[n][k], W row-major [N x K],
// ldw = K. Lanes 0-15: K = k0+{0..7, 16..23}, lanes 16-31: K = k0+{8..15, 24..31}; N = lane&15.
// => two 8-element bf16 runs per lane.
__device__ __forceinline__ v16bf load_bfrag(const bf16_t* __restrict__ W,
                                            int ldw, int n0, int k0, int lane) {
  int half = (lane >> 4) & 1;
  const bf16_t* p = W + (size_t)(n0 + (lane & 15)) * ldw + k0 + half * 8;
  v16bf b;
#pragma unroll
  for (int i = 0; i < 8; ++i) { b[i] = p[i]; b[i + 8] = p[16 + i]; }
  return b;
}

// C/D 16x16 f32 store: VGPR r, lanes 0-15 -> M = m0+r, lanes 16-31 -> M = m0+r+8.
__device__ __forceinline__ void store_c_f32(float* __restrict__ C, int ldc,
                                            int m0, int n0, int lane, v8f c,
                                            const float* __restrict__ bias,
                                            int ncols) {
  int half = (lane >> 4) & 1;
  int col = n0 + (lane & 15);
  if (col >= ncols) return;
  float bv = bias ? bias[col] : 0.0f;
#pragma unroll
  for (int r = 0; r < 8; ++r)
    C[(size_t)(m0 + r + half * 8) * ldc + col] = c[r] + bv;
}

// ---------------- small utility kernels ----------------
__global__ void k_f2bf(const float* __restrict__ s, bf16_t* __restrict__ d, int n) {
  int i = blockIdx.x * blockDim.x + threadIdx.x;
  if (i < n) d[i] = (bf16_t)s[i];
}

// enc_emb [V,E] fp32 -> embT [E][VPAD] bf16 (transposed, K-padded with zeros)
__global__ void k_trans_emb(const float* __restrict__ emb, bf16_t* __restrict__ embT) {
  int i = blockIdx.x * blockDim.x + threadIdx.x;  // E * VPAD
  if (i >= EDIM * VPAD) return;
  int e = i / VPAD, k = i % VPAD;
  embT[i] = (k < VDIM) ? (bf16_t)emb[(size_t)k * EDIM + e] : (bf16_t)0.0f;
}

// input [B,T,V] fp32 -> Xpad [(B*T)][VPAD] bf16 (zero pad)
__global__ void k_pad_input(const float* __restrict__ in, bf16_t* __restrict__ xp) {
  size_t i = (size_t)blockIdx.x * blockDim.x + threadIdx.x;
  if (i >= (size_t)BDIM * TDIM * VPAD) return;
  size_t row = i / VPAD; int k = (int)(i % VPAD);
  xp[i] = (k < VDIM) ? (bf16_t)in[row * VDIM + k] : (bf16_t)0.0f;
}

// outputs[:,0,:] = one_hot(START); max_output[:,0] = START; tok = START
__global__ void k_init(float* __restrict__ out0, float* __restrict__ out1,
                       int* __restrict__ tok) {
  int i = blockIdx.x * blockDim.x + threadIdx.x;
  if (i >= BDIM * VDIM) return;
  int b = i / VDIM, v = i % VDIM;
  out0[(size_t)b * TDIM * VDIM + v] = (v == START_TOK) ? 1.0f : 0.0f;
  if (v == 0) { out1[(size_t)b * TDIM] = (float)START_TOK; tok[b] = START_TOK; }
}

__global__ void k_zero_h(float* h1, float* h2, bf16_t* h1b, bf16_t* h2b) {
  int i = blockIdx.x * blockDim.x + threadIdx.x;
  if (i >= BDIM * HDIM) return;
  h1[i] = 0.0f; h2[i] = 0.0f; h1b[i] = (bf16_t)0.0f; h2b[i] = (bf16_t)0.0f;
}

// ---------------- big one-shot GEMM: X = input @ enc_emb  (bf16 WMMA) ----------
// A = Xpad [(B*T) x VPAD], B = embT [E x VPAD] (as W[N,K]); out -> Xbf [T][B][E]
__global__ void k_embed(const bf16_t* __restrict__ Xpad,
                        const bf16_t* __restrict__ embT,
                        bf16_t* __restrict__ Xbf) {
  int lane = threadIdx.x & 31, wave = threadIdx.x >> 5;
  int m0 = (blockIdx.x * 4 + wave) * 16;
  int n0 = blockIdx.y * 16;
  v8f c = {};
#pragma unroll
  for (int k0 = 0; k0 < VPAD; k0 += 32)
    c = wmma_bf16(load_afrag(Xpad, VPAD, m0, k0, lane),
                  load_bfrag(embT, VPAD, n0, k0, lane), c);
  int half = (lane >> 4) & 1;
  int col = n0 + (lane & 15);
#pragma unroll
  for (int r = 0; r < 8; ++r) {
    int rid = m0 + r + half * 8;               // rid = b*T + t
    int bb = rid >> 9, tt = rid & (TDIM - 1);
    Xbf[((size_t)tt * BDIM + bb) * EDIM + col] = (bf16_t)c[r];
  }
}

// ---------------- fused GRU gate GEMMs: gi = x@Wih^T + bih ; gh = h@Whh^T + bhh
// grid.x = 96 N-tiles of 3H; block = 128 (4 wave32 -> 4 M-tiles over batch 64)
__global__ void k_gates(const bf16_t* __restrict__ Xin, int Kx,
                        const bf16_t* __restrict__ Wih, const float* __restrict__ bih,
                        const bf16_t* __restrict__ Hbf,
                        const bf16_t* __restrict__ Whh, const float* __restrict__ bhh,
                        float* __restrict__ gi, float* __restrict__ gh) {
  int lane = threadIdx.x & 31, wave = threadIdx.x >> 5;
  int m0 = wave * 16;
  int n0 = blockIdx.x * 16;
  v8f ci = {}, ch = {};
  for (int k0 = 0; k0 < Kx; k0 += 32)
    ci = wmma_bf16(load_afrag(Xin, Kx, m0, k0, lane),
                   load_bfrag(Wih, Kx, n0, k0, lane), ci);
#pragma unroll
  for (int k0 = 0; k0 < HDIM; k0 += 32) {
    // prefetch next weight K-slice (gfx1250 global_prefetch_b8 path)
    if (k0 + 32 < HDIM)
      __builtin_prefetch(Whh + (size_t)(n0 + (lane & 15)) * HDIM + k0 + 32, 0, 1);
    ch = wmma_bf16(load_afrag(Hbf, HDIM, m0, k0, lane),
                   load_bfrag(Whh, HDIM, n0, k0, lane), ch);
  }
  store_c_f32(gi, G3H, m0, n0, lane, ci, bih, G3H);
  store_c_f32(gh, G3H, m0, n0, lane, ch, bhh, G3H);
}

// GRU pointwise: r,z,n gates -> h update (fp32 state + bf16 mirror + opt seq out)
__global__ void k_gru_pointwise(const float* __restrict__ gi, const float* __restrict__ gh,
                                float* __restrict__ h, bf16_t* __restrict__ hbf,
                                bf16_t* __restrict__ seq) {
  int i = blockIdx.x * blockDim.x + threadIdx.x;
  if (i >= BDIM * HDIM) return;
  int b = i >> 9, j = i & (HDIM - 1);
  size_t base = (size_t)b * G3H;
  float ir = gi[base + j],        hr = gh[base + j];
  float iz = gi[base + HDIM + j], hz = gh[base + HDIM + j];
  float in_ = gi[base + 2 * HDIM + j], hn = gh[base + 2 * HDIM + j];
  float r = 1.0f / (1.0f + expf(-(ir + hr)));
  float z = 1.0f / (1.0f + expf(-(iz + hz)));
  float n = tanhf(in_ + r * hn);
  float hv = (1.0f - z) * n + z * h[i];
  h[i] = hv; hbf[i] = (bf16_t)hv;
  if (seq) seq[i] = (bf16_t)hv;
}

// decoder token embedding gather -> xe bf16 [B,E]
__global__ void k_gather(const bf16_t* __restrict__ demb, const int* __restrict__ tok,
                         bf16_t* __restrict__ xe) {
  int i = blockIdx.x * blockDim.x + threadIdx.x;
  if (i >= BDIM * EDIM) return;
  int b = i >> 8, e = i & (EDIM - 1);
  xe[i] = demb[(size_t)tok[b] * EDIM + e];
}

// head layer 1: hid = LeakyReLU(h2 @ Wp1^T + bp1) -> bf16  (N=512 -> 32 tiles)
__global__ void k_head1(const bf16_t* __restrict__ h, const bf16_t* __restrict__ W,
                        const float* __restrict__ bias, bf16_t* __restrict__ out) {
  int lane = threadIdx.x & 31, wave = threadIdx.x >> 5;
  int m0 = wave * 16, n0 = blockIdx.x * 16;
  v8f c = {};
#pragma unroll
  for (int k0 = 0; k0 < HDIM; k0 += 32)
    c = wmma_bf16(load_afrag(h, HDIM, m0, k0, lane),
                  load_bfrag(W, HDIM, n0, k0, lane), c);
  int half = (lane >> 4) & 1;
  int col = n0 + (lane & 15);
#pragma unroll
  for (int r = 0; r < 8; ++r) {
    float v = c[r] + bias[col];
    v = (v >= 0.0f) ? v : 0.1f * v;
    out[(size_t)(m0 + r + half * 8) * HDIM + col] = (bf16_t)v;
  }
}

// head layer 2: logits = hid @ Wp2^T + bp2  (N=390 -> 25 tiles, guarded store)
__global__ void k_head2(const bf16_t* __restrict__ h, const bf16_t* __restrict__ W,
                        const float* __restrict__ bias, float* __restrict__ logits) {
  int lane = threadIdx.x & 31, wave = threadIdx.x >> 5;
  int m0 = wave * 16, n0 = blockIdx.x * 16;
  v8f c = {};
#pragma unroll
  for (int k0 = 0; k0 < HDIM; k0 += 32)
    c = wmma_bf16(load_afrag(h, HDIM, m0, k0, lane),
                  load_bfrag(W, HDIM, n0, k0, lane), c);
  store_c_f32(logits, VDIM, m0, n0, lane, c, bias, VDIM);
}

// softmax (probs -> outputs[:, t+1, :]) + gumbel-max sampling (hash RNG)
__device__ __forceinline__ unsigned hash3(unsigned a, unsigned b, unsigned c) {
  a += 0x9e3779b9u * b + 0x85ebca6bu * c;
  a ^= a >> 16; a *= 0x7feb352du; a ^= a >> 15; a *= 0x846ca68bu; a ^= a >> 16;
  return a;
}

__global__ void k_sample(const float* __restrict__ logits, float* __restrict__ out0,
                         float* __restrict__ out1, int* __restrict__ tok, int t) {
  int b = blockIdx.x, tid = threadIdx.x;     // 64 blocks x 128 threads
  __shared__ float sl[VDIM];
  __shared__ float red[128];
  __shared__ int redi[128];
  for (int v = tid; v < VDIM; v += 128) sl[v] = logits[(size_t)b * VDIM + v];
  __syncthreads();
  float m = -1e30f;
  for (int v = tid; v < VDIM; v += 128) m = fmaxf(m, sl[v]);
  red[tid] = m; __syncthreads();
  for (int s = 64; s > 0; s >>= 1) {
    if (tid < s) red[tid] = fmaxf(red[tid], red[tid + s]);
    __syncthreads();
  }
  float mx = red[0]; __syncthreads();
  float sum = 0.0f;
  for (int v = tid; v < VDIM; v += 128) sum += expf(sl[v] - mx);
  red[tid] = sum; __syncthreads();
  for (int s = 64; s > 0; s >>= 1) {
    if (tid < s) red[tid] += red[tid + s];
    __syncthreads();
  }
  float inv = 1.0f / red[0]; __syncthreads();
  float* dst = out0 + ((size_t)b * TDIM + (t + 1)) * VDIM;
  for (int v = tid; v < VDIM; v += 128) dst[v] = expf(sl[v] - mx) * inv;
  // gumbel-max sample
  float best = -1e30f; int bi = 0;
  for (int v = tid; v < VDIM; v += 128) {
    unsigned hsh = hash3(0x2545F491u ^ (unsigned)t, (unsigned)b, (unsigned)v);
    float u = ((hsh >> 8) + 0.5f) * (1.0f / 16777216.0f);
    float g = sl[v] - logf(-logf(u));
    if (g > best) { best = g; bi = v; }
  }
  red[tid] = best; redi[tid] = bi; __syncthreads();
  for (int s = 64; s > 0; s >>= 1) {
    if (tid < s && red[tid + s] > red[tid]) { red[tid] = red[tid + s]; redi[tid] = redi[tid + s]; }
    __syncthreads();
  }
  if (tid == 0) {
    tok[b] = redi[0];
    out1[(size_t)b * TDIM + (t + 1)] = (float)redi[0];
  }
}

// ---------------- host orchestration ----------------
extern "C" void kernel_launch(void* const* d_in, const int* in_sizes, int n_in,
                              void* d_out, int out_size, void* d_ws, size_t ws_size,
                              hipStream_t stream) {
  (void)in_sizes; (void)n_in; (void)out_size; (void)ws_size;

  // workspace carve-out (256B aligned)
  char* base = (char*)d_ws; size_t off = 0;
  auto A = [&](size_t bytes) -> void* {
    off = (off + 255) & ~(size_t)255;
    void* r = base + off; off += bytes; return r;
  };
  bf16_t* XPAD = (bf16_t*)A((size_t)BDIM * TDIM * VPAD * 2);
  bf16_t* XBF  = (bf16_t*)A((size_t)TDIM * BDIM * EDIM * 2);
  bf16_t* SEQ1 = (bf16_t*)A((size_t)TDIM * BDIM * HDIM * 2);
  bf16_t* EMBT = (bf16_t*)A((size_t)EDIM * VPAD * 2);
  bf16_t* WIH0E = (bf16_t*)A((size_t)G3H * EDIM * 2);
  bf16_t* WHH0E = (bf16_t*)A((size_t)G3H * HDIM * 2);
  bf16_t* WIH1E = (bf16_t*)A((size_t)G3H * HDIM * 2);
  bf16_t* WHH1E = (bf16_t*)A((size_t)G3H * HDIM * 2);
  bf16_t* WIH0D = (bf16_t*)A((size_t)G3H * EDIM * 2);
  bf16_t* WHH0D = (bf16_t*)A((size_t)G3H * HDIM * 2);
  bf16_t* WIH1D = (bf16_t*)A((size_t)G3H * HDIM * 2);
  bf16_t* WHH1D = (bf16_t*)A((size_t)G3H * HDIM * 2);
  bf16_t* WP1B  = (bf16_t*)A((size_t)HDIM * HDIM * 2);
  bf16_t* WP2B  = (bf16_t*)A((size_t)VDIM * HDIM * 2);
  bf16_t* DEMB  = (bf16_t*)A((size_t)VDIM * EDIM * 2);
  float*  H1  = (float*)A((size_t)BDIM * HDIM * 4);
  float*  H2  = (float*)A((size_t)BDIM * HDIM * 4);
  bf16_t* H1B = (bf16_t*)A((size_t)BDIM * HDIM * 2);
  bf16_t* H2B = (bf16_t*)A((size_t)BDIM * HDIM * 2);
  float*  GI  = (float*)A((size_t)BDIM * G3H * 4);
  float*  GH  = (float*)A((size_t)BDIM * G3H * 4);
  bf16_t* XE   = (bf16_t*)A((size_t)BDIM * EDIM * 2);
  bf16_t* HIDB = (bf16_t*)A((size_t)BDIM * HDIM * 2);
  float*  LOG  = (float*)A((size_t)BDIM * VDIM * 4);
  int*    TOK  = (int*)A((size_t)BDIM * 4);

  float* out0 = (float*)d_out;                         // outputs [B,T,V]
  float* out1 = out0 + (size_t)BDIM * TDIM * VDIM;     // max_output [B,T]

  auto cvt = [&](int idx, bf16_t* dst, int n) {
    k_f2bf<<<(n + 255) / 256, 256, 0, stream>>>((const float*)d_in[idx], dst, n);
  };
  // weight -> bf16 conversions (inputs per setup_inputs order)
  cvt(2,  WIH0E, G3H * EDIM);  cvt(3,  WHH0E, G3H * HDIM);
  cvt(6,  WIH1E, G3H * HDIM);  cvt(7,  WHH1E, G3H * HDIM);
  cvt(11, WIH0D, G3H * EDIM);  cvt(12, WHH0D, G3H * HDIM);
  cvt(15, WIH1D, G3H * HDIM);  cvt(16, WHH1D, G3H * HDIM);
  cvt(19, WP1B, HDIM * HDIM);  cvt(21, WP2B, VDIM * HDIM);
  cvt(10, DEMB, VDIM * EDIM);

  const float* bih0e = (const float*)d_in[4];  const float* bhh0e = (const float*)d_in[5];
  const float* bih1e = (const float*)d_in[8];  const float* bhh1e = (const float*)d_in[9];
  const float* bih0d = (const float*)d_in[13]; const float* bhh0d = (const float*)d_in[14];
  const float* bih1d = (const float*)d_in[16 + 1]; const float* bhh1d = (const float*)d_in[18];
  const float* bp1 = (const float*)d_in[20];   const float* bp2 = (const float*)d_in[22];

  {
    int n = EDIM * VPAD;
    k_trans_emb<<<(n + 255) / 256, 256, 0, stream>>>((const float*)d_in[1], EMBT);
  }
  {
    size_t n = (size_t)BDIM * TDIM * VPAD;
    k_pad_input<<<(unsigned)((n + 255) / 256), 256, 0, stream>>>((const float*)d_in[0], XPAD);
  }
  k_init<<<(BDIM * VDIM + 255) / 256, 256, 0, stream>>>(out0, out1, TOK);
  k_zero_h<<<(BDIM * HDIM + 255) / 256, 256, 0, stream>>>(H1, H2, H1B, H2B);

  // X = input @ enc_emb  (one big WMMA GEMM)
  k_embed<<<dim3(512, 16), 128, 0, stream>>>(XPAD, EMBT, XBF);

  // encoder layer 0 (h = H1), writes seq1
  for (int t = 0; t < TDIM; ++t) {
    k_gates<<<96, 128, 0, stream>>>(XBF + (size_t)t * BDIM * EDIM, EDIM,
                                    WIH0E, bih0e, H1B, WHH0E, bhh0e, GI, GH);
    k_gru_pointwise<<<(BDIM * HDIM + 255) / 256, 256, 0, stream>>>(
        GI, GH, H1, H1B, SEQ1 + (size_t)t * BDIM * HDIM);
  }
  // encoder layer 1 (h = H2 starts at zero)
  for (int t = 0; t < TDIM; ++t) {
    k_gates<<<96, 128, 0, stream>>>(SEQ1 + (size_t)t * BDIM * HDIM, HDIM,
                                    WIH1E, bih1e, H2B, WHH1E, bhh1e, GI, GH);
    k_gru_pointwise<<<(BDIM * HDIM + 255) / 256, 256, 0, stream>>>(
        GI, GH, H2, H2B, (bf16_t*)nullptr);
  }

  // decoder: H1/H2 continue from encoder final states
  for (int t = 0; t < TDIM - 1; ++t) {
    k_gather<<<(BDIM * EDIM + 255) / 256, 256, 0, stream>>>(DEMB, TOK, XE);
    k_gates<<<96, 128, 0, stream>>>(XE, EDIM, WIH0D, bih0d, H1B, WHH0D, bhh0d, GI, GH);
    k_gru_pointwise<<<(BDIM * HDIM + 255) / 256, 256, 0, stream>>>(
        GI, GH, H1, H1B, (bf16_t*)nullptr);
    k_gates<<<96, 128, 0, stream>>>(H1B, HDIM, WIH1D, bih1d, H2B, WHH1D, bhh1d, GI, GH);
    k_gru_pointwise<<<(BDIM * HDIM + 255) / 256, 256, 0, stream>>>(
        GI, GH, H2, H2B, (bf16_t*)nullptr);
    k_head1<<<32, 128, 0, stream>>>(H2B, WP1B, bp1, HIDB);
    k_head2<<<25, 128, 0, stream>>>(HIDB, WP2B, bp2, LOG);
    k_sample<<<BDIM, 128, 0, stream>>>(LOG, out0, out1, TOK, t);
  }
}
